// GATQueryProjector_9698036154645
// MI455X (gfx1250) — matrix operand order repaired and checked by reading, not compiled, not run on hardware
//
#include <hip/hip_runtime.h>
#include <cstdint>
#include <cstddef>

// ---------------------------------------------------------------------------
// 2-layer GAT for MI455X (gfx1250, wave32).
//  - bf16 WMMA GEMMs (v_wmma_f32_16x16x32_bf16), f32 accumulate, 32x64 wave tile
//  - dst-CSR built on device -> atomic-free segment softmax/aggregation
//  - async global->LDS staging (GLOBAL_LOAD_ASYNC_TO_LDS_B128) of h1[src] rows
//    overlapped with the softmax passes in the layer-1 aggregation
//  - layer-2 aggregation only for node 0 (reference returns h[0])
// ---------------------------------------------------------------------------

typedef __attribute__((ext_vector_type(16))) __bf16 bf16x16;
typedef __attribute__((ext_vector_type(8)))  __bf16 bf16x8;
typedef __attribute__((ext_vector_type(4)))  __bf16 bf16x4;
typedef __attribute__((ext_vector_type(8)))  float  f32x8;
typedef __attribute__((ext_vector_type(4)))  int    i32x4;

#define AS1 __attribute__((address_space(1)))
#define AS3 __attribute__((address_space(3)))

#if defined(__HIP_DEVICE_COMPILE__) && __has_builtin(__builtin_amdgcn_global_load_async_to_lds_b128)
#define GAT_ASYNC_LDS 1
#else
#define GAT_ASYNC_LDS 0
#endif

constexpr int IN_DIM = 768;
constexpr int HID    = 128;
constexpr int HEADS  = 4;
constexpr int OUT    = 128;
constexpr int HD1    = HEADS * HID;   // 512
constexpr int STAGE  = 16;            // edges staged in LDS per node (32KB)

static __device__ __forceinline__ float lrelu(float x) { return x > 0.f ? x : 0.2f * x; }

static __device__ __forceinline__ float wred_sum(float v) {
  #pragma unroll
  for (int o = 16; o > 0; o >>= 1) v += __shfl_xor(v, o, 32);
  return v;
}
static __device__ __forceinline__ float wred_max(float v) {
  #pragma unroll
  for (int o = 16; o > 0; o >>= 1) v = fmaxf(v, __shfl_xor(v, o, 32));
  return v;
}

// ---------------------------------------------------------------------------
// f32 -> bf16 cast (vectorized, grid-stride)
// ---------------------------------------------------------------------------
__global__ __launch_bounds__(256)
void cast_f32_bf16(const float* __restrict__ in, __bf16* __restrict__ out, size_t n)
{
  const size_t stride = (size_t)gridDim.x * blockDim.x;
  for (size_t i = (size_t)blockIdx.x * blockDim.x + threadIdx.x; i * 4 < n; i += stride) {
    const size_t k = i * 4;
    if (k + 4 <= n) {
      const float4 v = *(const float4*)(in + k);
      bf16x4 o;
      o[0] = (__bf16)v.x; o[1] = (__bf16)v.y; o[2] = (__bf16)v.z; o[3] = (__bf16)v.w;
      *(bf16x4*)(out + k) = o;
    } else {
      for (size_t j = k; j < n; ++j) out[j] = (__bf16)in[j];
    }
  }
}

// W[K][Nc] f32  ->  WT[Nc][K] bf16   (small matrices; done once per launch)
__global__ __launch_bounds__(256)
void transpose_cast(const float* __restrict__ W, __bf16* __restrict__ WT, int K, int Nc)
{
  const int idx = blockIdx.x * 256 + threadIdx.x;
  if (idx < K * Nc) {
    const int n = idx / K;
    const int k = idx - n * K;
    WT[idx] = (__bf16)W[(size_t)k * Nc + n];
  }
}

// ---------------------------------------------------------------------------
// C[M][Nc] f32 = A[M][K] bf16 x BT[Nc][K] bf16   (BT is B transposed)
// One wave per block; wave owns a 32x64 output tile:
//   2 A-fragments x 4 B-fragments -> 8 WMMAs per K-step of 32
//   (12 b128 loads / 8 WMMAs = 1.5 loads per matrix op).
//
// Fragment layouts (CDNA5 ISA 7.12.2, 16-bit):
//   A 16x32 : lane<16 -> row=lane, K {0..15}; lane>=16 -> K {16..31} (contiguous K chosen here, not the ISA order)
//             => two contiguous 16B loads per lane from the A row.
//   B 32x16 : VGPR j packs K=2j,2j+1; lanes 0..15 K 0..15, lanes 16..31 K 16..31 (same order as A, not ISA's)
//             => with BT row-major-along-K this is ONE contiguous 32B load per lane.
//   C 16x16 : VGPR r -> (M=r, N=lane) lanes 0..15, (M=r+8, N=lane-16) lanes 16..31.
// ---------------------------------------------------------------------------
static __device__ __forceinline__ bf16x16 load_afrag(const __bf16* arow, int k0, int half)
{
  const bf16x8 alo = *(const bf16x8*)(arow + k0 + half * 16);
  const bf16x8 ahi = *(const bf16x8*)(arow + k0 + half * 16 + 8);
  bf16x16 a;
  #pragma unroll
  for (int i = 0; i < 8; ++i) { a[i] = alo[i]; a[i + 8] = ahi[i]; }
  return a;
}

__global__ __launch_bounds__(32)
void gemm_wmma_bf16(const __bf16* __restrict__ A, const __bf16* __restrict__ BT,
                    float* __restrict__ C, int M, int Nc, int K)
{
  const int lane = threadIdx.x;
  const int half = lane >> 4;      // 0 or 1
  const int l16  = lane & 15;
  const int m0   = blockIdx.x * 32;
  const int n0   = blockIdx.y * 64;

  const int mr0 = (m0 + l16      < M) ? (m0 + l16)      : (M - 1);  // tail-safe
  const int mr1 = (m0 + 16 + l16 < M) ? (m0 + 16 + l16) : (M - 1);
  const __bf16* __restrict__ arow0 = A + (size_t)mr0 * K;
  const __bf16* __restrict__ arow1 = A + (size_t)mr1 * K;
  const __bf16* __restrict__ bbas  = BT + (size_t)(n0 + l16) * K + half * 16;
  const size_t bs16 = (size_t)16 * K;

  f32x8 c00 = {0,0,0,0,0,0,0,0}, c01 = {0,0,0,0,0,0,0,0};
  f32x8 c02 = {0,0,0,0,0,0,0,0}, c03 = {0,0,0,0,0,0,0,0};
  f32x8 c10 = {0,0,0,0,0,0,0,0}, c11 = {0,0,0,0,0,0,0,0};
  f32x8 c12 = {0,0,0,0,0,0,0,0}, c13 = {0,0,0,0,0,0,0,0};

  for (int k0 = 0; k0 < K; k0 += 32) {
    const bf16x16 a0 = load_afrag(arow0, k0, half);
    const bf16x16 a1 = load_afrag(arow1, k0, half);

    const bf16x16 b0 = *(const bf16x16*)(bbas + k0);
    const bf16x16 b1 = *(const bf16x16*)(bbas + k0 + bs16);
    const bf16x16 b2 = *(const bf16x16*)(bbas + k0 + 2 * bs16);
    const bf16x16 b3 = *(const bf16x16*)(bbas + k0 + 3 * bs16);

    if (k0 + 32 < K) {
      __builtin_prefetch((const void*)(arow0 + k0 + 32), 0, 1);
      __builtin_prefetch((const void*)(arow1 + k0 + 32), 0, 1);
    }

    c00 = __builtin_amdgcn_wmma_f32_16x16x32_bf16(false, a0, false, b0, (short)0, c00, false, false);
    c01 = __builtin_amdgcn_wmma_f32_16x16x32_bf16(false, a0, false, b1, (short)0, c01, false, false);
    c02 = __builtin_amdgcn_wmma_f32_16x16x32_bf16(false, a0, false, b2, (short)0, c02, false, false);
    c03 = __builtin_amdgcn_wmma_f32_16x16x32_bf16(false, a0, false, b3, (short)0, c03, false, false);
    c10 = __builtin_amdgcn_wmma_f32_16x16x32_bf16(false, a1, false, b0, (short)0, c10, false, false);
    c11 = __builtin_amdgcn_wmma_f32_16x16x32_bf16(false, a1, false, b1, (short)0, c11, false, false);
    c12 = __builtin_amdgcn_wmma_f32_16x16x32_bf16(false, a1, false, b2, (short)0, c12, false, false);
    c13 = __builtin_amdgcn_wmma_f32_16x16x32_bf16(false, a1, false, b3, (short)0, c13, false, false);
  }

  #pragma unroll
  for (int r = 0; r < 8; ++r) {
    const int row0 = m0 + half * 8 + r;
    const int row1 = row0 + 16;
    if (row0 < M) {
      float* cr = C + (size_t)row0 * Nc + n0 + l16;
      cr[0] = c00[r]; cr[16] = c01[r]; cr[32] = c02[r]; cr[48] = c03[r];
    }
    if (row1 < M) {
      float* cr = C + (size_t)row1 * Nc + n0 + l16;
      cr[0] = c10[r]; cr[16] = c11[r]; cr[32] = c12[r]; cr[48] = c13[r];
    }
  }
}

// ---------------------------------------------------------------------------
// alpha_s[n][h] = dot(h[n,h,:], a_src[h,:]) ; alpha_d likewise. D == 128.
// One wave per node.
// ---------------------------------------------------------------------------
__global__ __launch_bounds__(32)
void alpha_kernel(const float* __restrict__ h, const float* __restrict__ aw_s,
                  const float* __restrict__ aw_d, float* __restrict__ as_out,
                  float* __restrict__ ad_out, int H)
{
  const int n = blockIdx.x, lane = threadIdx.x;
  for (int hd = 0; hd < H; ++hd) {
    const float4 v = *(const float4*)(h + ((size_t)n * H + hd) * 128 + lane * 4);
    const float4 s = *(const float4*)(aw_s + (size_t)hd * 128 + lane * 4);
    const float4 d = *(const float4*)(aw_d + (size_t)hd * 128 + lane * 4);
    float ps = v.x * s.x + v.y * s.y + v.z * s.z + v.w * s.w;
    float pd = v.x * d.x + v.y * d.y + v.z * d.z + v.w * d.w;
    ps = wred_sum(ps);
    pd = wred_sum(pd);
    if (lane == 0) {
      as_out[(size_t)n * H + hd] = ps;
      ad_out[(size_t)n * H + hd] = pd;
    }
  }
}

// ---------------------------------------------------------------------------
// CSR build over virtual edge list [0,E) = edge_index, [E,E+N) = self loops
// ---------------------------------------------------------------------------
__global__ __launch_bounds__(256)
void zero_i32(int* __restrict__ p, int n)
{
  const int t = blockIdx.x * 256 + threadIdx.x;
  if (t < n) p[t] = 0;
}

__global__ __launch_bounds__(256)
void copy_i32(const int* __restrict__ a, int* __restrict__ b, int n)
{
  const int t = blockIdx.x * 256 + threadIdx.x;
  if (t < n) b[t] = a[t];
}

__global__ __launch_bounds__(256)
void count_edges(const int* __restrict__ ei, int E, int N, int* __restrict__ counts)
{
  const int t = blockIdx.x * 256 + threadIdx.x;
  if (t < E + N) {
    const int d = (t < E) ? ei[E + t] : (t - E);
    atomicAdd(&counts[d], 1);
  }
}

__global__ __launch_bounds__(256)
void scan_phase1(const int* __restrict__ counts, int* __restrict__ bsum, int N)
{
  __shared__ int sh[256];
  const int b = blockIdx.x, t = threadIdx.x;
  const int base = b * 1024 + t * 4;
  int s = 0;
  #pragma unroll
  for (int j = 0; j < 4; ++j) { const int idx = base + j; if (idx < N) s += counts[idx]; }
  sh[t] = s;
  __syncthreads();
  for (int o = 128; o > 0; o >>= 1) { if (t < o) sh[t] += sh[t + o]; __syncthreads(); }
  if (t == 0) bsum[b] = sh[0];
}

__global__ void scan_phase2(int* __restrict__ bsum, int NB)
{
  if (threadIdx.x == 0 && blockIdx.x == 0) {
    int run = 0;
    for (int i = 0; i < NB; ++i) { const int v = bsum[i]; bsum[i] = run; run += v; }
  }
}

__global__ __launch_bounds__(256)
void scan_phase3(const int* __restrict__ counts, const int* __restrict__ bsum,
                 int* __restrict__ offs, int N)
{
  __shared__ int sh[256];
  const int b = blockIdx.x, t = threadIdx.x;
  const int base = b * 1024 + t * 4;
  int v[4];
  int tot = 0;
  #pragma unroll
  for (int j = 0; j < 4; ++j) { v[j] = (base + j < N) ? counts[base + j] : 0; tot += v[j]; }
  sh[t] = tot;
  __syncthreads();
  for (int o = 1; o < 256; o <<= 1) {
    const int mine  = sh[t];
    const int other = (t >= o) ? sh[t - o] : 0;
    __syncthreads();
    sh[t] = mine + other;
    __syncthreads();
  }
  int prefix = bsum[b] + sh[t] - tot;   // exclusive prefix of this thread's 4 items
  #pragma unroll
  for (int j = 0; j < 4; ++j) {
    if (base + j < N) {
      offs[base + j] = prefix;
      prefix += v[j];
      if (base + j == N - 1) offs[N] = prefix;
    }
  }
}

__global__ __launch_bounds__(256)
void fill_csr(const int* __restrict__ ei, int E, int N,
              int* __restrict__ cursor, int* __restrict__ csrc)
{
  const int t = blockIdx.x * 256 + threadIdx.x;
  if (t < E + N) {
    int s, d;
    if (t < E) { s = ei[t]; d = ei[E + t]; }
    else       { s = t - E; d = t - E; }
    const int p = atomicAdd(&cursor[d], 1);
    csrc[p] = s;
  }
}

// ---------------------------------------------------------------------------
// Layer-1 per-destination softmax + aggregation + bias + ReLU -> x2 (bf16).
// One wave per node. 512 channels = 4 heads x (32 lanes x float4).
// The h1[src] rows of the first STAGE edges are DMA'd into LDS with
// GLOBAL_LOAD_ASYNC_TO_LDS_B128 *before* the softmax passes, hiding the gather
// latency behind passes 1-2; s_wait_asynccnt 0 before consuming them.
// ---------------------------------------------------------------------------
__global__ __launch_bounds__(32)
void aggregate_l1(const float* __restrict__ h1, const float* __restrict__ as1,
                  const float* __restrict__ ad1, const int* __restrict__ offs,
                  const int* __restrict__ csrc, const float* __restrict__ bias,
                  __bf16* __restrict__ x2)
{
  const int n = blockIdx.x;
  const int lane = threadIdx.x;
  const int base = offs[n];
  const int deg  = offs[n + 1] - base;
  const float4 ad = *(const float4*)(ad1 + (size_t)n * 4);

#if GAT_ASYNC_LDS
  __shared__ __align__(16) float lbuf[STAGE * HD1];   // 32 KB
  const int nst = (deg < STAGE) ? deg : STAGE;
  for (int i = 0; i < nst; ++i) {
    const int s = csrc[base + i];
    const float* gsrc = h1 + (size_t)s * HD1 + lane * 4;
    float*       ldst = lbuf + (size_t)i * HD1 + lane * 4;
    #pragma unroll
    for (int c = 0; c < 4; ++c) {
      __builtin_amdgcn_global_load_async_to_lds_b128(
          (AS1 i32x4*)(unsigned long long)(const void*)(gsrc + c * 128),
          (AS3 i32x4*)(ldst + c * 128), 0, 0);
    }
  }
#else
  const int nst = 0;
#endif

  // pass 1: per-head max over incoming edges (overlaps with the async DMA)
  float m0 = -3.402823466e38f, m1 = m0, m2 = m0, m3 = m0;
  for (int i = lane; i < deg; i += 32) {
    const int s = csrc[base + i];
    const float4 av = *(const float4*)(as1 + (size_t)s * 4);
    m0 = fmaxf(m0, lrelu(av.x + ad.x));
    m1 = fmaxf(m1, lrelu(av.y + ad.y));
    m2 = fmaxf(m2, lrelu(av.z + ad.z));
    m3 = fmaxf(m3, lrelu(av.w + ad.w));
  }
  m0 = wred_max(m0); m1 = wred_max(m1); m2 = wred_max(m2); m3 = wred_max(m3);

  // pass 2: sum of exp
  float s0 = 0.f, s1 = 0.f, s2 = 0.f, s3 = 0.f;
  for (int i = lane; i < deg; i += 32) {
    const int s = csrc[base + i];
    const float4 av = *(const float4*)(as1 + (size_t)s * 4);
    s0 += __expf(lrelu(av.x + ad.x) - m0);
    s1 += __expf(lrelu(av.y + ad.y) - m1);
    s2 += __expf(lrelu(av.z + ad.z) - m2);
    s3 += __expf(lrelu(av.w + ad.w) - m3);
  }
  s0 = wred_sum(s0); s1 = wred_sum(s1); s2 = wred_sum(s2); s3 = wred_sum(s3);
  const float r0 = 1.f / (s0 + 1e-16f);
  const float r1 = 1.f / (s1 + 1e-16f);
  const float r2 = 1.f / (s2 + 1e-16f);
  const float r3 = 1.f / (s3 + 1e-16f);

#if GAT_ASYNC_LDS
  asm volatile("s_wait_asynccnt 0x0" ::: "memory");
#endif

  // pass 3: weighted aggregation of h1[src] (LDS-staged rows first, then global)
  float4 a0 = {0,0,0,0}, a1 = {0,0,0,0}, a2 = {0,0,0,0}, a3 = {0,0,0,0};
  for (int i = 0; i < deg; ++i) {
    const int s = csrc[base + i];
    const float4 av = *(const float4*)(as1 + (size_t)s * 4);
    const float w0 = __expf(lrelu(av.x + ad.x) - m0) * r0;
    const float w1 = __expf(lrelu(av.y + ad.y) - m1) * r1;
    const float w2 = __expf(lrelu(av.z + ad.z) - m2) * r2;
    const float w3 = __expf(lrelu(av.w + ad.w) - m3) * r3;
    float4 v0, v1, v2, v3;
#if GAT_ASYNC_LDS
    if (i < nst) {
      const float4* lr = (const float4*)(lbuf + (size_t)i * HD1);
      v0 = lr[lane]; v1 = lr[32 + lane]; v2 = lr[64 + lane]; v3 = lr[96 + lane];
    } else
#endif
    {
      const float4* hr = (const float4*)(h1 + (size_t)s * HD1);
      v0 = hr[lane]; v1 = hr[32 + lane]; v2 = hr[64 + lane]; v3 = hr[96 + lane];
    }
    a0.x += w0*v0.x; a0.y += w0*v0.y; a0.z += w0*v0.z; a0.w += w0*v0.w;
    a1.x += w1*v1.x; a1.y += w1*v1.y; a1.z += w1*v1.z; a1.w += w1*v1.w;
    a2.x += w2*v2.x; a2.y += w2*v2.y; a2.z += w2*v2.z; a2.w += w2*v2.w;
    a3.x += w3*v3.x; a3.y += w3*v3.y; a3.z += w3*v3.z; a3.w += w3*v3.w;
  }

  // epilogue: + bias, ReLU, store bf16 (feeds GEMM2 directly)
  const float4 b0 = *(const float4*)(bias + 0 * 128 + lane * 4);
  const float4 b1 = *(const float4*)(bias + 1 * 128 + lane * 4);
  const float4 b2 = *(const float4*)(bias + 2 * 128 + lane * 4);
  const float4 b3 = *(const float4*)(bias + 3 * 128 + lane * 4);
  __bf16* xrow = x2 + (size_t)n * HD1;
  bf16x4 o;
  o[0]=(__bf16)fmaxf(a0.x+b0.x,0.f); o[1]=(__bf16)fmaxf(a0.y+b0.y,0.f);
  o[2]=(__bf16)fmaxf(a0.z+b0.z,0.f); o[3]=(__bf16)fmaxf(a0.w+b0.w,0.f);
  *(bf16x4*)(xrow + 0 * 128 + lane * 4) = o;
  o[0]=(__bf16)fmaxf(a1.x+b1.x,0.f); o[1]=(__bf16)fmaxf(a1.y+b1.y,0.f);
  o[2]=(__bf16)fmaxf(a1.z+b1.z,0.f); o[3]=(__bf16)fmaxf(a1.w+b1.w,0.f);
  *(bf16x4*)(xrow + 1 * 128 + lane * 4) = o;
  o[0]=(__bf16)fmaxf(a2.x+b2.x,0.f); o[1]=(__bf16)fmaxf(a2.y+b2.y,0.f);
  o[2]=(__bf16)fmaxf(a2.z+b2.z,0.f); o[3]=(__bf16)fmaxf(a2.w+b2.w,0.f);
  *(bf16x4*)(xrow + 2 * 128 + lane * 4) = o;
  o[0]=(__bf16)fmaxf(a3.x+b3.x,0.f); o[1]=(__bf16)fmaxf(a3.y+b3.y,0.f);
  o[2]=(__bf16)fmaxf(a3.z+b3.z,0.f); o[3]=(__bf16)fmaxf(a3.w+b3.w,0.f);
  *(bf16x4*)(xrow + 3 * 128 + lane * 4) = o;
}

// ---------------------------------------------------------------------------
// Layer-2 aggregation for node 0 only (reference output is h[0]).
// H=1, D=128; writes d_out directly (+ bias; mean over 1 head == identity).
// ---------------------------------------------------------------------------
__global__ __launch_bounds__(32)
void aggregate_l2_node0(const float* __restrict__ h2, const float* __restrict__ as2,
                        const float* __restrict__ ad2, const int* __restrict__ offs,
                        const int* __restrict__ csrc, const float* __restrict__ b2,
                        float* __restrict__ out)
{
  const int lane = threadIdx.x;
  const int base = offs[0];
  const int deg  = offs[1] - offs[0];
  const float adv = ad2[0];

  float mx = -3.402823466e38f;
  for (int i = lane; i < deg; i += 32) {
    const int s = csrc[base + i];
    mx = fmaxf(mx, lrelu(as2[s] + adv));
  }
  mx = wred_max(mx);

  float sm = 0.f;
  for (int i = lane; i < deg; i += 32) {
    const int s = csrc[base + i];
    sm += __expf(lrelu(as2[s] + adv) - mx);
  }
  sm = wred_sum(sm);
  const float rd = 1.f / (sm + 1e-16f);

  float ax = 0.f, ay = 0.f, az = 0.f, aw = 0.f;
  for (int i = 0; i < deg; ++i) {
    const int s = csrc[base + i];
    const float w = __expf(lrelu(as2[s] + adv) - mx) * rd;
    const float4 v = *(const float4*)(h2 + (size_t)s * OUT + lane * 4);
    ax += w * v.x; ay += w * v.y; az += w * v.z; aw += w * v.w;
  }
  const float4 bb = *(const float4*)(b2 + lane * 4);
  out[lane * 4 + 0] = ax + bb.x;
  out[lane * 4 + 1] = ay + bb.y;
  out[lane * 4 + 2] = az + bb.z;
  out[lane * 4 + 3] = aw + bb.w;
}

// ---------------------------------------------------------------------------
// Host-side orchestration (all async on `stream`; graph-capture safe)
// ---------------------------------------------------------------------------
extern "C" void kernel_launch(void* const* d_in, const int* in_sizes, int n_in,
                              void* d_out, int out_size, void* d_ws, size_t ws_size,
                              hipStream_t stream)
{
  (void)n_in; (void)out_size; (void)ws_size;

  const float* x    = (const float*)d_in[0];
  const int*   ei   = (const int*)  d_in[1];
  const float* W1   = (const float*)d_in[2];
  const float* aS1  = (const float*)d_in[3];
  const float* aD1  = (const float*)d_in[4];
  const float* b1   = (const float*)d_in[5];
  const float* W2   = (const float*)d_in[6];
  const float* aS2  = (const float*)d_in[7];
  const float* aD2  = (const float*)d_in[8];
  const float* b2   = (const float*)d_in[9];
  float* out = (float*)d_out;

  const int N  = in_sizes[0] / IN_DIM;   // 50000
  const int E  = in_sizes[1] / 2;        // 500000
  const int EV = E + N;                  // + self loops

  // ---- workspace carve (256B aligned) -----------------------------------
  uint8_t* ws  = (uint8_t*)d_ws;
  size_t   off = 0;
  auto carve = [&](size_t bytes) -> uint8_t* {
    uint8_t* p = ws + off;
    off = (off + bytes + 255) & ~(size_t)255;
    return p;
  };
  __bf16* xbf    = (__bf16*)carve((size_t)N * IN_DIM * sizeof(__bf16)); // 76.8MB (reused as x2)
  float*  h1     = (float*) carve((size_t)N * HD1 * sizeof(float));     // 102.4MB (reused as h2)
  __bf16* w1t    = (__bf16*)carve((size_t)HD1 * IN_DIM * sizeof(__bf16));
  __bf16* w2t    = (__bf16*)carve((size_t)OUT * HD1 * sizeof(__bf16));
  float*  as1    = (float*) carve((size_t)N * HEADS * sizeof(float));
  float*  ad1    = (float*) carve((size_t)N * HEADS * sizeof(float));
  float*  as2    = (float*) carve((size_t)N * sizeof(float));
  float*  ad2    = (float*) carve((size_t)N * sizeof(float));
  int*    counts = (int*)   carve((size_t)N * sizeof(int));
  int*    offs   = (int*)   carve(((size_t)N + 1) * sizeof(int));
  int*    cursor = (int*)   carve((size_t)N * sizeof(int));
  int*    bsum   = (int*)   carve(4096);
  int*    csrc   = (int*)   carve((size_t)EV * sizeof(int));
  __bf16* x2 = xbf;   // X-bf16 is dead once GEMM1 finished
  float*  h2 = h1;    // h1 is dead once layer-1 aggregation finished

  // ---- 1) precision/layout prep -----------------------------------------
  {
    const size_t nx = (size_t)N * IN_DIM;
    const int blocks = (int)((nx / 4 + 255) / 256);
    cast_f32_bf16<<<blocks, 256, 0, stream>>>(x, xbf, nx);
  }
  transpose_cast<<<(IN_DIM * HD1 + 255) / 256, 256, 0, stream>>>(W1, w1t, IN_DIM, HD1);
  transpose_cast<<<(HD1 * OUT + 255) / 256, 256, 0, stream>>>(W2, w2t, HD1, OUT);

  // ---- 2) GEMM1 + alpha1 -------------------------------------------------
  gemm_wmma_bf16<<<dim3((N + 31) / 32, HD1 / 64), 32, 0, stream>>>(xbf, w1t, h1, N, HD1, IN_DIM);
  alpha_kernel<<<N, 32, 0, stream>>>(h1, aS1, aD1, as1, ad1, HEADS);

  // ---- 3) CSR build (dst-sorted incoming edges, incl. self loops) --------
  zero_i32<<<(N + 255) / 256, 256, 0, stream>>>(counts, N);
  count_edges<<<(EV + 255) / 256, 256, 0, stream>>>(ei, E, N, counts);
  const int NB = (N + 1023) / 1024;
  scan_phase1<<<NB, 256, 0, stream>>>(counts, bsum, N);
  scan_phase2<<<1, 32, 0, stream>>>(bsum, NB);
  scan_phase3<<<NB, 256, 0, stream>>>(counts, bsum, offs, N);
  copy_i32<<<(N + 255) / 256, 256, 0, stream>>>(offs, cursor, N);
  fill_csr<<<(EV + 255) / 256, 256, 0, stream>>>(ei, E, N, cursor, csrc);

  // ---- 4) layer-1 softmax/aggregate -> x2 (bf16) --------------------------
  aggregate_l1<<<N, 32, 0, stream>>>(h1, as1, ad1, offs, csrc, b1, x2);

  // ---- 5) GEMM2 + alpha2 --------------------------------------------------
  gemm_wmma_bf16<<<dim3((N + 31) / 32, OUT / 64), 32, 0, stream>>>(x2, w2t, h2, N, OUT, HD1);
  alpha_kernel<<<N, 32, 0, stream>>>(h2, aS2, aD2, as2, ad2, 1);

  // ---- 6) layer-2 aggregation, node 0 only -> d_out -----------------------
  aggregate_l2_node0<<<1, 32, 0, stream>>>(h2, as2, ad2, offs, csrc, b2, out);
}
